// MixedAttentionProcessor_14173392076864
// MI455X (gfx1250) — matrix operand order repaired and checked by reading, not compiled, not run on hardware
//
#include <hip/hip_runtime.h>
#include <hip/hip_bf16.h>

// ---------------- problem constants ----------------
#define DIMC     512
#define NHEADS   8
#define HDIM     64
#define SEQ      4096
#define ATT_SCALE 0.125f // 64^-0.5

// ---------------- WMMA types ----------------
typedef __attribute__((ext_vector_type(16))) __bf16 v16bf;
typedef __attribute__((ext_vector_type(8)))  float  v8f;

union Frag16 { uint4 u[2]; v16bf v; };

__device__ __forceinline__ unsigned short f2bf(float f) {
  __bf16 h = (__bf16)f;                       // native bf16 cvt (RNE)
  return __builtin_bit_cast(unsigned short, h);
}
__device__ __forceinline__ unsigned int pack2(float a, float b) {
  return (unsigned int)f2bf(a) | ((unsigned int)f2bf(b) << 16);
}

// ---------------- LDS layout (ushort units) ----------------
#define XS_STR 528                     // 512 + 16 pad, rows 1056B (16B aligned)
#define VT_STR 72                      // 64 + 8 pad,  rows 144B  (16B aligned)
#define PS_STR 72
#define XS_OFF 0                       // x window  64 x 512 bf16 (reused as O)
#define QS_OFF (64 * XS_STR)           // q  64 tok x 512
#define KS_OFF (2 * 64 * XS_STR)       // k  64 tok x 512
#define VT_OFF (3 * 64 * XS_STR)       // v^T 512 hd x 64 tok
#define PS_OFF (VT_OFF + 512 * VT_STR) // softmax P scratch: 16 waves x 16 x 64
#define OS_OFF 0
#define SMEM_USHORTS (PS_OFF + 16 * 16 * PS_STR)  // 156672 -> 313344 B < 320KB

// ---------------- fragment loaders ----------------
// A-frag 16x32 bf16: lane row = l%16; 16 contiguous K at g*16
__device__ __forceinline__ v16bf lds_afrag(const unsigned short* s, int off,
                                           int stride, int row, int kOff, int g) {
  const unsigned short* p = s + off + row * stride + kOff + g * 16;
  Frag16 f;
  f.u[0] = *(const uint4*)p;
  f.u[1] = *(const uint4*)(p + 8);
  return f.v;
}
// B-frag 32x16 bf16 from Bt (N-major): lane col = l%16; 16 contiguous K at g*16
__device__ __forceinline__ v16bf lds_bfrag(const unsigned short* s, int off,
                                           int stride, int n, int kOff, int g) {
  const unsigned short* p = s + off + n * stride + kOff + g * 16;
  Frag16 f;
  f.u[0] = *(const uint4*)p;
  f.u[1] = *(const uint4*)(p + 8);
  return f.v;
}
__device__ __forceinline__ v16bf gbl_bfrag(const unsigned short* bt, int stride,
                                           int n, int kOff, int g) {
  const unsigned short* p = bt + (size_t)n * stride + kOff + g * 16;
  Frag16 f;
  f.u[0] = *(const uint4*)p;
  f.u[1] = *(const uint4*)(p + 8);
  return f.v;
}

#define WMMA_BF16(A, B, C) \
  __builtin_amdgcn_wmma_f32_16x16x32_bf16(false, (A), false, (B), (short)0, (C), false, false)

// ---------------- weight prep: transpose + f32->bf16 ----------------
__global__ void prep_weights(const float* __restrict__ qkv_w,
                             const float* __restrict__ proj_w,
                             unsigned short* __restrict__ wqkvT,
                             unsigned short* __restrict__ wprojT) {
  int idx = blockIdx.x * 256 + threadIdx.x;       // exactly 1,048,576 threads
  if (idx < 1536 * 512) {
    int n = idx / 512, k = idx % 512;             // wqkvT[n][k] = qkv_w[k][n]
    wqkvT[idx] = f2bf(qkv_w[(size_t)k * 1536 + n]);
  } else {
    int j = idx - 1536 * 512;
    int n = j / 512, k = j % 512;                 // wprojT[n][k] = proj_w[k][n]
    wprojT[j] = f2bf(proj_w[(size_t)k * 512 + n]);
  }
}

// ---------------- fused per-window attention (16 waves / window) ----------------
__global__ __launch_bounds__(512, 1)
void swin_attn_fused(const float* __restrict__ hidden,     // [16,4096,512]
                     const float* __restrict__ qkv_b,      // [1536]
                     const float* __restrict__ proj_b,     // [512]
                     const unsigned short* __restrict__ wqkvT,  // [1536][512] bf16
                     const unsigned short* __restrict__ wprojT, // [512][512]  bf16
                     float* __restrict__ out) {            // [16,4096,512]
  extern __shared__ unsigned short sm[];
  const int tid  = threadIdx.x;
  const int lane = tid & 31;
  const int wv   = tid >> 5;          // wave 0..15
  const int g    = lane >> 4;         // half-wave group
  const int ln   = lane & 15;

  const int blk = blockIdx.x;         // 0..1023 = batch*64 + window
  const int b   = blk >> 6;
  const int win = blk & 63;
  const int wy  = win >> 3, wx = win & 7;
  const int baseGT = wy * 512 + wx * 8;   // (wy*8)*64 + wx*8

  // ===== Phase 1: gather x window -> LDS bf16 (window partition) =====
  {
    const int tok  = tid >> 3;        // 0..63
    const int part = tid & 7;         // 64 floats each
    const int gt   = baseGT + (tok >> 3) * 64 + (tok & 7);
    const float4* src = (const float4*)(hidden + (size_t)(b * SEQ + gt) * DIMC) + part * 16;
    unsigned short* dst = sm + XS_OFF + tok * XS_STR + part * 64;
#pragma unroll
    for (int i = 0; i < 16; ++i) {
      float4 f = src[i];
      uint2 u;
      u.x = pack2(f.x, f.y);
      u.y = pack2(f.z, f.w);
      *(uint2*)(dst + i * 4) = u;
    }
  }
  __syncthreads();

  // ===== Phase 2: QKV GEMM (64 x 1536, K=512), scatter q/k/v^T to LDS =====
  // Each wave: 6 n-tiles as 2 groups of 3; A-frags explicitly cached per k-step,
  // kk loop kept rolled so nothing gets LICM-hoisted into spills.
#pragma unroll
  for (int grp = 0; grp < 2; ++grp) {
    v8f acc[3][4] = {};
#pragma unroll 1
    for (int kk = 0; kk < 16; ++kk) {
      v16bf af[4];
#pragma unroll
      for (int mt = 0; mt < 4; ++mt)
        af[mt] = lds_afrag(sm, XS_OFF, XS_STR, mt * 16 + ln, kk * 32, g);
#pragma unroll
      for (int j = 0; j < 3; ++j) {
        const int nt = wv + 16 * (3 * grp + j);
        v16bf bf = gbl_bfrag(wqkvT, 512, nt * 16 + ln, kk * 32, g);
#pragma unroll
        for (int mt = 0; mt < 4; ++mt)
          acc[j][mt] = WMMA_BF16(af[mt], bf, acc[j][mt]);
      }
    }
#pragma unroll
    for (int j = 0; j < 3; ++j) {
      const int nt = wv + 16 * (3 * grp + j);
      const int jc = nt * 16 + ln;    // output column (uniform segment per tile)
      const float bias = qkv_b[jc];
      if (jc < 512) {                 // ---- Q: token-major
#pragma unroll
        for (int mt = 0; mt < 4; ++mt)
#pragma unroll
          for (int r = 0; r < 8; ++r)
            sm[QS_OFF + (mt * 16 + g * 8 + r) * XS_STR + jc] = f2bf(acc[j][mt][r] + bias);
      } else if (jc < 1024) {         // ---- K: token-major
        const int c = jc - 512;
#pragma unroll
        for (int mt = 0; mt < 4; ++mt)
#pragma unroll
          for (int r = 0; r < 8; ++r)
            sm[KS_OFF + (mt * 16 + g * 8 + r) * XS_STR + c] = f2bf(acc[j][mt][r] + bias);
      } else {                        // ---- V: transposed (hd-major), packed b32 stores
        const int c = jc - 1024;
#pragma unroll
        for (int mt = 0; mt < 4; ++mt) {
          unsigned short* vp = sm + VT_OFF + c * VT_STR + mt * 16 + g * 8;
#pragma unroll
          for (int i = 0; i < 4; ++i)
            *(unsigned int*)(vp + 2 * i) =
                pack2(acc[j][mt][2 * i] + bias, acc[j][mt][2 * i + 1] + bias);
        }
      }
    }
  }
  __syncthreads();

  // ===== Phase 3: attention; 2 waves per head, 2 m-strips per wave =====
  {
    const int h     = wv >> 1;
    const int mt0   = (wv & 1) * 2;
    const int qBase = QS_OFF + h * HDIM;          // column shift into q
    const int kBase = KS_OFF + h * HDIM;
    const int vBase = VT_OFF + h * HDIM * VT_STR; // row shift into v^T
    const int pBase = PS_OFF + wv * 16 * PS_STR;  // wave-private P scratch

#pragma unroll 1
    for (int ms = 0; ms < 2; ++ms) {
      const int mt = mt0 + ms;
      // ---- S strip (16 x 64) = q_strip @ k^T, f32 accum
      v8f sacc[4] = {};
#pragma unroll
      for (int kk = 0; kk < 2; ++kk) {
        v16bf qa = lds_afrag(sm, qBase, XS_STR, mt * 16 + ln, kk * 32, g);
#pragma unroll
        for (int nt = 0; nt < 4; ++nt) {
          v16bf kb = lds_bfrag(sm, kBase, XS_STR, nt * 16 + ln, kk * 32, g);
          sacc[nt] = WMMA_BF16(qa, kb, sacc[nt]);
        }
      }
      // ---- row softmax: row (g*8+r) spans the 16 lanes of this half-group
#pragma unroll
      for (int r = 0; r < 8; ++r) {
        float s0 = sacc[0][r], s1 = sacc[1][r], s2 = sacc[2][r], s3 = sacc[3][r];
        float mx = fmaxf(fmaxf(s0, s1), fmaxf(s2, s3));
        mx = fmaxf(mx, __shfl_xor(mx, 1, 32));
        mx = fmaxf(mx, __shfl_xor(mx, 2, 32));
        mx = fmaxf(mx, __shfl_xor(mx, 4, 32));
        mx = fmaxf(mx, __shfl_xor(mx, 8, 32));
        float e0 = __expf((s0 - mx) * ATT_SCALE);
        float e1 = __expf((s1 - mx) * ATT_SCALE);
        float e2 = __expf((s2 - mx) * ATT_SCALE);
        float e3 = __expf((s3 - mx) * ATT_SCALE);
        float sum = e0 + e1 + e2 + e3;
        sum += __shfl_xor(sum, 1, 32);
        sum += __shfl_xor(sum, 2, 32);
        sum += __shfl_xor(sum, 4, 32);
        sum += __shfl_xor(sum, 8, 32);
        float inv = __builtin_amdgcn_rcpf(sum);
        unsigned short* pr = sm + pBase + (g * 8 + r) * PS_STR + ln;
        pr[0]  = f2bf(e0 * inv);
        pr[16] = f2bf(e1 * inv);
        pr[32] = f2bf(e2 * inv);
        pr[48] = f2bf(e3 * inv);
      }
      // ---- O strip (16 x 64) = P_strip @ V  (B-frags from v^T)
      v16bf pa0 = lds_afrag(sm, pBase, PS_STR, ln, 0,  g);
      v16bf pa1 = lds_afrag(sm, pBase, PS_STR, ln, 32, g);
#pragma unroll
      for (int nt = 0; nt < 4; ++nt) {
        v8f oacc = {};
        v16bf vb0 = lds_bfrag(sm, vBase, VT_STR, nt * 16 + ln, 0,  g);
        v16bf vb1 = lds_bfrag(sm, vBase, VT_STR, nt * 16 + ln, 32, g);
        oacc = WMMA_BF16(pa0, vb0, oacc);
        oacc = WMMA_BF16(pa1, vb1, oacc);
        const int col = h * HDIM + nt * 16 + ln;
#pragma unroll
        for (int r = 0; r < 8; ++r)
          sm[OS_OFF + (mt * 16 + g * 8 + r) * XS_STR + col] = f2bf(oacc[r]);
      }
    }
  }
  __syncthreads();

  // ===== Phase 4: proj GEMM (64 x 512, K=512) + bias + window reverse =====
  {
    v8f acc[2][4] = {};
#pragma unroll 1
    for (int kk = 0; kk < 16; ++kk) {
      v16bf af[4];
#pragma unroll
      for (int mt = 0; mt < 4; ++mt)
        af[mt] = lds_afrag(sm, OS_OFF, XS_STR, mt * 16 + ln, kk * 32, g);
#pragma unroll
      for (int j = 0; j < 2; ++j) {
        const int nt = wv + 16 * j;
        v16bf bf = gbl_bfrag(wprojT, 512, nt * 16 + ln, kk * 32, g);
#pragma unroll
        for (int mt = 0; mt < 4; ++mt)
          acc[j][mt] = WMMA_BF16(af[mt], bf, acc[j][mt]);
      }
    }
#pragma unroll
    for (int j = 0; j < 2; ++j) {
      const int nt  = wv + 16 * j;
      const int col = nt * 16 + ln;
      const float pb = proj_b[col];
#pragma unroll
      for (int mt = 0; mt < 4; ++mt)
#pragma unroll
        for (int r = 0; r < 8; ++r) {
          const int tok = mt * 16 + g * 8 + r;
          const int gt  = baseGT + (tok >> 3) * 64 + (tok & 7);
          out[(size_t)(b * SEQ + gt) * DIMC + col] = acc[j][mt][r] + pb;
        }
    }
  }
}

// ---------------- host entry ----------------
extern "C" void kernel_launch(void* const* d_in, const int* in_sizes, int n_in,
                              void* d_out, int out_size, void* d_ws, size_t ws_size,
                              hipStream_t stream) {
  // inputs: [0]=attn_output(unused) [1]=hidden_states [2]=qkv_w [3]=qkv_b [4]=proj_w [5]=proj_b
  const float* hidden = (const float*)d_in[1];
  const float* qkv_w  = (const float*)d_in[2];
  const float* qkv_b  = (const float*)d_in[3];
  const float* proj_w = (const float*)d_in[4];
  const float* proj_b = (const float*)d_in[5];

  unsigned short* wqkvT  = (unsigned short*)d_ws;          // 1536*512 bf16
  unsigned short* wprojT = wqkvT + 1536 * 512;             // 512*512 bf16

  prep_weights<<<4096, 256, 0, stream>>>(qkv_w, proj_w, wqkvT, wprojT);

  const size_t smem = SMEM_USHORTS * sizeof(unsigned short); // 313,344 B
  swin_attn_fused<<<1024, 512, smem, stream>>>(hidden, qkv_b, proj_b,
                                               wqkvT, wprojT, (float*)d_out);
}
